// GraphAttentionNet_45878840656040
// MI455X (gfx1250) — compile-verified, matrix operand order checked by reading, not run on hardware
//
#include <hip/hip_runtime.h>

#define NN 384
#define DF 128

typedef __attribute__((ext_vector_type(16))) __bf16 bf16x16;
typedef __attribute__((ext_vector_type(8)))  float  f32x8;

union FragU { bf16x16 v; int4 q[2]; };

__device__ inline __bf16 f2bf(float x) {
    union { float f; unsigned u; } cv; cv.f = x;
    unsigned u = cv.u;
    u += 0x7FFFu + ((u >> 16) & 1u);
    unsigned short h = (unsigned short)(u >> 16);
    union { unsigned short s; __bf16 b; } o; o.s = h;
    return o.b;
}
__device__ inline float bf2f(__bf16 b) {
    union { unsigned short s; __bf16 b; } iu; iu.b = b;
    union { unsigned u; float f; } o; o.u = ((unsigned)iu.s) << 16;
    return o.f;
}
__device__ inline float lrelu(float v) { return v >= 0.f ? v : 0.2f * v; }

// 16-bit A/B operand fragment: lane L holds row (L&15), K = half*8+0..7 and 16+half*8+0..7
__device__ inline bf16x16 load_frag(const __bf16* rowp, int k0) {
    const int half = (threadIdx.x >> 4) & 1;
    const __bf16* p = rowp + k0 + half * 8;
    FragU f;
    f.q[0] = *(const int4*)(p);
    f.q[1] = *(const int4*)(p + 16);
    return f.v;
}
__device__ inline bf16x16 zero_frag() {
    FragU f;
    f.q[0] = make_int4(0, 0, 0, 0);
    f.q[1] = make_int4(0, 0, 0, 0);
    return f.v;
}
__device__ inline f32x8 wmma_bf16(bf16x16 a, bf16x16 b, f32x8 c) {
    return __builtin_amdgcn_wmma_f32_16x16x32_bf16(false, a, false, b, (short)0, c, false, false);
}

// ---------------- f32 -> bf16 convert ----------------
__global__ void cvt_kernel(const float* __restrict__ s, __bf16* __restrict__ d, int n) {
    for (int idx = blockIdx.x * blockDim.x + threadIdx.x; idx < n; idx += gridDim.x * blockDim.x)
        d[idx] = f2bf(s[idx]);
}

// ---------------- fused per-layer pair kernel ----------------
// grid (24,24), 128 threads (4 waves), 128KB dynamic LDS.
// x_pw row (i,j) = [ xn[j] | xn[i] | xe[i,j] | xe[j,i] ], K=512.
// Computes att logits (pre-softmax) and the new edge features (edge MLP + residual).
__global__ __launch_bounds__(128) void pair_kernel(
    const __bf16* __restrict__ xn, const __bf16* __restrict__ xe,
    const __bf16* __restrict__ aw1, const float* __restrict__ ab1,
    const __bf16* __restrict__ aw2, const float* __restrict__ ab2,
    const __bf16* __restrict__ ew1, const float* __restrict__ eb1,
    const __bf16* __restrict__ ew2, const float* __restrict__ eb2,
    float* __restrict__ logits, __bf16* __restrict__ xe_out)
{
    extern __shared__ __bf16 lds[];     // phase1/2: h1[256][256]; phase3: wave-private he
    const int tid  = threadIdx.x;
    const int lane = tid & 31, wave = tid >> 5;
    const int r    = lane & 15, half = lane >> 4;
    const int i0   = blockIdx.y * 16, j0 = blockIdx.x * 16;
    const int iB   = i0 + wave * 4;
    const int j    = j0 + r;

    // Phase 1: h1 = lrelu(x_pw @ aw1^T + ab1), K=512, N=256 (wave-private rows of LDS)
    #pragma unroll 1
    for (int nc = 0; nc < 4; ++nc) {
        f32x8 acc[4][4] = {};
        #pragma unroll
        for (int seg = 0; seg < 4; ++seg) {          // unrolled: pointer select is compile-time
            const __bf16* rp[4];
            #pragma unroll
            for (int m = 0; m < 4; ++m) {
                const int i = iB + m;
                rp[m] = (seg == 0) ? xn + (size_t)j * DF
                      : (seg == 1) ? xn + (size_t)i * DF
                      : (seg == 2) ? xe + ((size_t)i * NN + j) * DF
                                   : xe + ((size_t)j * NN + i) * DF;
            }
            #pragma unroll 1
            for (int ks = 0; ks < 4; ++ks) {
                const int kk = ks * 32;
                const int k0 = seg * 128 + kk;
                bf16x16 bfr[4], afr[4];
                #pragma unroll
                for (int t = 0; t < 4; ++t)
                    bfr[t] = load_frag(aw1 + (size_t)((nc * 4 + t) * 16 + r) * 512, k0);
                #pragma unroll
                for (int m = 0; m < 4; ++m)
                    afr[m] = load_frag(rp[m], kk);
                #pragma unroll
                for (int m = 0; m < 4; ++m) {
                    #pragma unroll
                    for (int t = 0; t < 4; ++t)
                        acc[m][t] = wmma_bf16(afr[m], bfr[t], acc[m][t]);
                }
            }
        }
        #pragma unroll
        for (int m = 0; m < 4; ++m) {
            const int pb = (wave * 4 + m) * 16 + half * 8;
            #pragma unroll
            for (int t = 0; t < 4; ++t) {
                const int n = (nc * 4 + t) * 16 + r;
                const float bias = ab1[n];
                #pragma unroll
                for (int e = 0; e < 8; ++e)
                    lds[(size_t)(pb + e) * 256 + n] = f2bf(lrelu(acc[m][t][e] + bias));
            }
        }
    }

    // Phase 2: logits = lrelu(h1 @ aw2^T + ab2), K=256, N=4 (padded to 16)
    #pragma unroll 1
    for (int m = 0; m < 4; ++m) {
        const int mt = wave * 4 + m;
        f32x8 acc = {};
        #pragma unroll 1
        for (int ks = 0; ks < 8; ++ks) {
            const int k0 = ks * 32;
            bf16x16 b = (r < 4) ? load_frag(aw2 + (size_t)r * 256, k0) : zero_frag();
            bf16x16 a = load_frag(lds + (size_t)(mt * 16 + r) * 256, k0);
            acc = wmma_bf16(a, b, acc);
        }
        if (r < 4) {
            const float bias = ab2[r];
            const int i = i0 + mt;
            #pragma unroll
            for (int e = 0; e < 8; ++e) {
                const int jj = j0 + half * 8 + e;
                logits[((size_t)i * NN + jj) * 4 + r] = lrelu(acc[e] + bias);
            }
        }
    }

    // Phase 3a: he = lrelu(x_pw @ ew1^T + eb1), K=512, N=128 (wave-private LDS region)
    __bf16* he = lds + (size_t)wave * 16384;   // 64 rows x 128
    #pragma unroll 1
    for (int nc = 0; nc < 2; ++nc) {
        f32x8 acc[4][4] = {};
        #pragma unroll
        for (int seg = 0; seg < 4; ++seg) {
            const __bf16* rp[4];
            #pragma unroll
            for (int m = 0; m < 4; ++m) {
                const int i = iB + m;
                rp[m] = (seg == 0) ? xn + (size_t)j * DF
                      : (seg == 1) ? xn + (size_t)i * DF
                      : (seg == 2) ? xe + ((size_t)i * NN + j) * DF
                                   : xe + ((size_t)j * NN + i) * DF;
            }
            #pragma unroll 1
            for (int ks = 0; ks < 4; ++ks) {
                const int kk = ks * 32;
                const int k0 = seg * 128 + kk;
                bf16x16 bfr[4], afr[4];
                #pragma unroll
                for (int t = 0; t < 4; ++t)
                    bfr[t] = load_frag(ew1 + (size_t)((nc * 4 + t) * 16 + r) * 512, k0);
                #pragma unroll
                for (int m = 0; m < 4; ++m)
                    afr[m] = load_frag(rp[m], kk);
                #pragma unroll
                for (int m = 0; m < 4; ++m) {
                    #pragma unroll
                    for (int t = 0; t < 4; ++t)
                        acc[m][t] = wmma_bf16(afr[m], bfr[t], acc[m][t]);
                }
            }
        }
        #pragma unroll
        for (int m = 0; m < 4; ++m) {
            const int pl = m * 16 + half * 8;
            #pragma unroll
            for (int t = 0; t < 4; ++t) {
                const int n = (nc * 4 + t) * 16 + r;
                const float bias = eb1[n];
                #pragma unroll
                for (int e = 0; e < 8; ++e)
                    he[(size_t)(pl + e) * 128 + n] = f2bf(lrelu(acc[m][t][e] + bias));
            }
        }
    }

    // Phase 3b: xe_out = lrelu(he @ ew2^T + eb2) + xe
    #pragma unroll 1
    for (int nc = 0; nc < 2; ++nc) {
        f32x8 acc[4][4] = {};
        #pragma unroll 1
        for (int ks = 0; ks < 4; ++ks) {
            const int k0 = ks * 32;
            bf16x16 bfr[4];
            #pragma unroll
            for (int t = 0; t < 4; ++t)
                bfr[t] = load_frag(ew2 + (size_t)((nc * 4 + t) * 16 + r) * 128, k0);
            #pragma unroll
            for (int m = 0; m < 4; ++m) {
                bf16x16 a = load_frag(he + (size_t)(m * 16 + r) * 128, k0);
                #pragma unroll
                for (int t = 0; t < 4; ++t) acc[m][t] = wmma_bf16(a, bfr[t], acc[m][t]);
            }
        }
        #pragma unroll
        for (int m = 0; m < 4; ++m) {
            const int i = i0 + wave * 4 + m;
            #pragma unroll
            for (int t = 0; t < 4; ++t) {
                const int n = (nc * 4 + t) * 16 + r;
                const float bias = eb2[n];
                #pragma unroll
                for (int e = 0; e < 8; ++e) {
                    const int jj = j0 + half * 8 + e;
                    const size_t off = ((size_t)i * NN + jj) * DF + n;
                    xe_out[off] = f2bf(lrelu(acc[m][t][e] + bias) + bf2f(xe[off]));
                }
            }
        }
    }
}

// ---------------- softmax over j (axis 1), in place, [NN][NN][4] ----------------
__global__ __launch_bounds__(128) void softmax_kernel(float* __restrict__ lg) {
    const int i = blockIdx.x;
    __shared__ float red[128];
    for (int h = 0; h < 4; ++h) {
        float mx = -1e30f;
        for (int jj = threadIdx.x; jj < NN; jj += 128)
            mx = fmaxf(mx, lg[((size_t)i * NN + jj) * 4 + h]);
        red[threadIdx.x] = mx; __syncthreads();
        for (int s = 64; s > 0; s >>= 1) {
            if (threadIdx.x < s) red[threadIdx.x] = fmaxf(red[threadIdx.x], red[threadIdx.x + s]);
            __syncthreads();
        }
        mx = red[0]; __syncthreads();
        float sm = 0.f;
        for (int jj = threadIdx.x; jj < NN; jj += 128) {
            const size_t o = ((size_t)i * NN + jj) * 4 + h;
            const float e = __expf(lg[o] - mx);
            lg[o] = e; sm += e;
        }
        red[threadIdx.x] = sm; __syncthreads();
        for (int s = 64; s > 0; s >>= 1) {
            if (threadIdx.x < s) red[threadIdx.x] += red[threadIdx.x + s];
            __syncthreads();
        }
        const float inv = 1.f / red[0]; __syncthreads();
        for (int jj = threadIdx.x; jj < NN; jj += 128)
            lg[((size_t)i * NN + jj) * 4 + h] *= inv;
    }
}

// ---------------- x_out[i, h*512+d] = sum_j att[i,j,h] * x_pw[i,j,d] ----------------
__global__ __launch_bounds__(256) void einsum_kernel(
    const float* __restrict__ att, const __bf16* __restrict__ xn,
    const __bf16* __restrict__ xe, __bf16* __restrict__ x_out)
{
    const int i = blockIdx.x;
    const int d = threadIdx.x * 2;
    const int seg = d >> 7, kk = d & 127;
    // hoist source base/stride: p(jj) = base + jj*stride
    const __bf16* base;
    size_t stride;
    if (seg == 0)      { base = xn + kk;                            stride = DF; }
    else if (seg == 1) { base = xn + (size_t)i * DF + kk;           stride = 0;  }
    else if (seg == 2) { base = xe + ((size_t)i * NN) * DF + kk;    stride = DF; }
    else               { base = xe + (size_t)i * DF + kk;           stride = (size_t)NN * DF; }

    float a0 = 0.f, a1 = 0.f, b0 = 0.f, b1 = 0.f, c0 = 0.f, c1 = 0.f, d0 = 0.f, d1 = 0.f;
    const float* ap = att + (size_t)i * NN * 4;
    #pragma unroll 1
    for (int jj = 0; jj < NN; ++jj) {
        const float4 w = *(const float4*)(ap + (size_t)jj * 4);
        const __bf16* p = base + (size_t)jj * stride;
        const float v0 = bf2f(p[0]), v1 = bf2f(p[1]);
        a0 += w.x * v0; a1 += w.x * v1;
        b0 += w.y * v0; b1 += w.y * v1;
        c0 += w.z * v0; c1 += w.z * v1;
        d0 += w.w * v0; d1 += w.w * v1;
    }
    __bf16* o = x_out + (size_t)i * 2048 + d;
    o[0]    = f2bf(a0); o[1]    = f2bf(a1);
    o[512]  = f2bf(b0); o[513]  = f2bf(b1);
    o[1024] = f2bf(c0); o[1025] = f2bf(c1);
    o[1536] = f2bf(d0); o[1537] = f2bf(d1);
}

// ---------------- node MLP: xn_new = lrelu(mlp(x_out)) + xn ----------------
__global__ __launch_bounds__(128) void node_mlp_kernel(
    const __bf16* __restrict__ x_out, const __bf16* __restrict__ w1, const float* __restrict__ b1,
    const __bf16* __restrict__ w2, const float* __restrict__ b2,
    const __bf16* __restrict__ xn_in, __bf16* __restrict__ xn_out)
{
    __shared__ __bf16 hl[16 * 128];
    const int tid = threadIdx.x, lane = tid & 31, wave = tid >> 5;
    const int r = lane & 15, half = lane >> 4;
    const int i0 = blockIdx.x * 16;

    f32x8 acc[2] = {};
    #pragma unroll 1
    for (int ks = 0; ks < 64; ++ks) {
        const int k0 = ks * 32;
        bf16x16 a = load_frag(x_out + (size_t)(i0 + r) * 2048, k0);
        #pragma unroll
        for (int t = 0; t < 2; ++t) {
            bf16x16 b = load_frag(w1 + (size_t)((wave * 2 + t) * 16 + r) * 2048, k0);
            acc[t] = wmma_bf16(a, b, acc[t]);
        }
    }
    #pragma unroll
    for (int t = 0; t < 2; ++t) {
        const int n = (wave * 2 + t) * 16 + r;
        const float bias = b1[n];
        #pragma unroll
        for (int e = 0; e < 8; ++e)
            hl[(half * 8 + e) * 128 + n] = f2bf(lrelu(acc[t][e] + bias));
    }
    __syncthreads();
    f32x8 a2[2] = {};
    #pragma unroll 1
    for (int ks = 0; ks < 4; ++ks) {
        const int k0 = ks * 32;
        bf16x16 a = load_frag(hl + (size_t)r * 128, k0);
        #pragma unroll
        for (int t = 0; t < 2; ++t) {
            bf16x16 b = load_frag(w2 + (size_t)((wave * 2 + t) * 16 + r) * 128, k0);
            a2[t] = wmma_bf16(a, b, a2[t]);
        }
    }
    #pragma unroll
    for (int t = 0; t < 2; ++t) {
        const int n = (wave * 2 + t) * 16 + r;
        const float bias = b2[n];
        #pragma unroll
        for (int e = 0; e < 8; ++e) {
            const int i = i0 + half * 8 + e;
            const float v = lrelu(a2[t][e] + bias) + bf2f(xn_in[(size_t)i * 128 + n]);
            xn_out[(size_t)i * 128 + n] = f2bf(v);
        }
    }
}

// ---------------- final node transform: 128 -> 128 -> 64 (last linear) ----------------
__global__ __launch_bounds__(128) void final_node_kernel(
    const __bf16* __restrict__ xn, const __bf16* __restrict__ w1, const float* __restrict__ b1,
    const __bf16* __restrict__ w2, const float* __restrict__ b2, float* __restrict__ out)
{
    __shared__ __bf16 hl[16 * 128];
    const int tid = threadIdx.x, lane = tid & 31, wave = tid >> 5;
    const int r = lane & 15, half = lane >> 4;
    const int i0 = blockIdx.x * 16;

    f32x8 acc[2] = {};
    #pragma unroll 1
    for (int ks = 0; ks < 4; ++ks) {
        const int k0 = ks * 32;
        bf16x16 a = load_frag(xn + (size_t)(i0 + r) * 128, k0);
        #pragma unroll
        for (int t = 0; t < 2; ++t) {
            bf16x16 b = load_frag(w1 + (size_t)((wave * 2 + t) * 16 + r) * 128, k0);
            acc[t] = wmma_bf16(a, b, acc[t]);
        }
    }
    #pragma unroll
    for (int t = 0; t < 2; ++t) {
        const int n = (wave * 2 + t) * 16 + r;
        const float bias = b1[n];
        #pragma unroll
        for (int e = 0; e < 8; ++e)
            hl[(half * 8 + e) * 128 + n] = f2bf(lrelu(acc[t][e] + bias));
    }
    __syncthreads();
    f32x8 a2 = {};
    #pragma unroll 1
    for (int ks = 0; ks < 4; ++ks) {
        const int k0 = ks * 32;
        bf16x16 a = load_frag(hl + (size_t)r * 128, k0);
        bf16x16 b = load_frag(w2 + (size_t)(wave * 16 + r) * 128, k0);
        a2 = wmma_bf16(a, b, a2);
    }
    const int n = wave * 16 + r;
    const float bias = b2[n];
    #pragma unroll
    for (int e = 0; e < 8; ++e) {
        const int i = i0 + half * 8 + e;
        out[(size_t)i * 64 + n] = a2[e] + bias;     // last_linear: no lrelu
    }
}

// ---------------- final edge transform: 128 -> 128 -> 32 (last linear), M = NN*NN ----------------
__global__ __launch_bounds__(128) void final_edge_kernel(
    const __bf16* __restrict__ xe, const __bf16* __restrict__ w1, const float* __restrict__ b1,
    const __bf16* __restrict__ w2, const float* __restrict__ b2, float* __restrict__ out)
{
    __shared__ __bf16 hl[64 * 128];
    const int tid = threadIdx.x, lane = tid & 31, wave = tid >> 5;
    const int r = lane & 15, half = lane >> 4;
    const int rb = blockIdx.x * 64 + wave * 16;
    __bf16* hw = hl + (size_t)wave * 16 * 128;     // wave-private rows

    bf16x16 afr[4];
    #pragma unroll
    for (int ks = 0; ks < 4; ++ks)
        afr[ks] = load_frag(xe + (size_t)(rb + r) * 128, ks * 32);

    #pragma unroll 1
    for (int nc = 0; nc < 2; ++nc) {
        f32x8 acc[4] = {};
        #pragma unroll 1
        for (int ks = 0; ks < 4; ++ks) {
            const int k0 = ks * 32;
            #pragma unroll
            for (int t = 0; t < 4; ++t) {
                bf16x16 b = load_frag(w1 + (size_t)((nc * 4 + t) * 16 + r) * 128, k0);
                acc[t] = wmma_bf16(afr[ks], b, acc[t]);
            }
        }
        #pragma unroll
        for (int t = 0; t < 4; ++t) {
            const int n = (nc * 4 + t) * 16 + r;
            const float bias = b1[n];
            #pragma unroll
            for (int e = 0; e < 8; ++e)
                hw[(half * 8 + e) * 128 + n] = f2bf(lrelu(acc[t][e] + bias));
        }
    }
    f32x8 a2[2] = {};
    #pragma unroll 1
    for (int ks = 0; ks < 4; ++ks) {
        const int k0 = ks * 32;
        bf16x16 a = load_frag(hw + (size_t)r * 128, k0);
        #pragma unroll
        for (int t = 0; t < 2; ++t) {
            bf16x16 b = load_frag(w2 + (size_t)(t * 16 + r) * 128, k0);
            a2[t] = wmma_bf16(a, b, a2[t]);
        }
    }
    #pragma unroll
    for (int t = 0; t < 2; ++t) {
        const int n = t * 16 + r;
        const float bias = b2[n];
        #pragma unroll
        for (int e = 0; e < 8; ++e) {
            const int row = rb + half * 8 + e;
            out[(size_t)row * 32 + n] = a2[t][e] + bias;   // last_linear
        }
    }
}

// ==================== host ====================
extern "C" void kernel_launch(void* const* d_in, const int* in_sizes, int n_in,
                              void* d_out, int out_size, void* d_ws, size_t ws_size,
                              hipStream_t stream) {
    (void)in_sizes; (void)n_in; (void)out_size; (void)ws_size;

    const float* feat_node = (const float*)d_in[0];
    const float* feat_edge = (const float*)d_in[1];
    const float *aw1f[3], *ab1f[3], *aw2f[3], *ab2f[3];
    const float *nw1f[3], *nb1f[3], *nw2f[3], *nb2f[3];
    const float *ew1f[3], *eb1f[3], *ew2f[3], *eb2f[3];
    for (int l = 0; l < 3; ++l) {
        const int base = 2 + l * 12;
        aw1f[l] = (const float*)d_in[base + 0];  ab1f[l] = (const float*)d_in[base + 1];
        aw2f[l] = (const float*)d_in[base + 2];  ab2f[l] = (const float*)d_in[base + 3];
        nw1f[l] = (const float*)d_in[base + 4];  nb1f[l] = (const float*)d_in[base + 5];
        nw2f[l] = (const float*)d_in[base + 6];  nb2f[l] = (const float*)d_in[base + 7];
        ew1f[l] = (const float*)d_in[base + 8];  eb1f[l] = (const float*)d_in[base + 9];
        ew2f[l] = (const float*)d_in[base + 10]; eb2f[l] = (const float*)d_in[base + 11];
    }
    const float* tw1f = (const float*)d_in[38]; const float* tb1f = (const float*)d_in[39];
    const float* tw2f = (const float*)d_in[40]; const float* tb2f = (const float*)d_in[41];
    const float* fw1f = (const float*)d_in[42]; const float* fb1f = (const float*)d_in[43];
    const float* fw2f = (const float*)d_in[44]; const float* fb2f = (const float*)d_in[45];

    char* wp = (char*)d_ws;
    auto carve = [&](size_t bytes) -> char* {
        char* p = wp;
        wp += (bytes + 255) & ~(size_t)255;
        return p;
    };
    __bf16* xe_b[2] = { (__bf16*)carve((size_t)NN * NN * DF * 2),
                        (__bf16*)carve((size_t)NN * NN * DF * 2) };
    __bf16* xn_b[2] = { (__bf16*)carve((size_t)NN * DF * 2),
                        (__bf16*)carve((size_t)NN * DF * 2) };
    float*  logits = (float*)carve((size_t)NN * NN * 4 * 4);
    __bf16* x_out  = (__bf16*)carve((size_t)NN * 2048 * 2);
    __bf16 *aw1[3], *aw2[3], *nw1[3], *nw2[3], *ew1[3], *ew2[3];
    for (int l = 0; l < 3; ++l) {
        aw1[l] = (__bf16*)carve(256 * 512 * 2);
        aw2[l] = (__bf16*)carve(4 * 256 * 2);
        nw1[l] = (__bf16*)carve(128 * 2048 * 2);
        nw2[l] = (__bf16*)carve(128 * 128 * 2);
        ew1[l] = (__bf16*)carve(128 * 512 * 2);
        ew2[l] = (__bf16*)carve(128 * 128 * 2);
    }
    __bf16* tw1 = (__bf16*)carve(128 * 128 * 2);
    __bf16* tw2 = (__bf16*)carve(64 * 128 * 2);
    __bf16* fw1 = (__bf16*)carve(128 * 128 * 2);
    __bf16* fw2 = (__bf16*)carve(32 * 128 * 2);

    auto cvt = [&](const float* s, __bf16* d, size_t n) {
        int blocks = (int)((n + 255) / 256);
        if (blocks > 16384) blocks = 16384;
        cvt_kernel<<<blocks, 256, 0, stream>>>(s, d, (int)n);
    };
    cvt(feat_node, xn_b[0], (size_t)NN * DF);
    cvt(feat_edge, xe_b[0], (size_t)NN * NN * DF);
    for (int l = 0; l < 3; ++l) {
        cvt(aw1f[l], aw1[l], 256 * 512);
        cvt(aw2f[l], aw2[l], 4 * 256);
        cvt(nw1f[l], nw1[l], 128 * 2048);
        cvt(nw2f[l], nw2[l], 128 * 128);
        cvt(ew1f[l], ew1[l], 128 * 512);
        cvt(ew2f[l], ew2[l], 128 * 128);
    }
    cvt(tw1f, tw1, 128 * 128);
    cvt(tw2f, tw2, 64 * 128);
    cvt(fw1f, fw1, 128 * 128);
    cvt(fw2f, fw2, 32 * 128);

    int cur = 0;
    for (int l = 0; l < 3; ++l) {
        const int nxt = cur ^ 1;
        pair_kernel<<<dim3(24, 24), 128, 131072, stream>>>(
            xn_b[cur], xe_b[cur],
            aw1[l], ab1f[l], aw2[l], ab2f[l],
            ew1[l], eb1f[l], ew2[l], eb2f[l],
            logits, xe_b[nxt]);
        softmax_kernel<<<NN, 128, 0, stream>>>(logits);
        einsum_kernel<<<NN, 256, 0, stream>>>(logits, xn_b[cur], xe_b[cur], x_out);
        node_mlp_kernel<<<NN / 16, 128, 0, stream>>>(
            x_out, nw1[l], nb1f[l], nw2[l], nb2f[l], xn_b[cur], xn_b[nxt]);
        cur = nxt;
    }
    final_node_kernel<<<NN / 16, 128, 0, stream>>>(
        xn_b[cur], tw1, tb1f, tw2, tb2f, (float*)d_out);
    final_edge_kernel<<<(NN * NN) / 64, 128, 0, stream>>>(
        xe_b[cur], fw1, fb1f, fw2, fb2f, (float*)d_out + (size_t)NN * 64);
}
